// FieldInteractionOperators_25331717112425
// MI455X (gfx1250) — matrix operand order repaired and checked by reading, not compiled, not run on hardware
//
#include <hip/hip_runtime.h>
#include <hip/hip_bf16.h>

// Problem constants (from reference): B=4, N=8192, D=512, H=8, hd=64
#define BB 4
#define NN 8192
#define DD 512
#define MT 32            // rows per workgroup tile
#define SP 520           // padded LDS row stride in bf16 elems (260 dwords = 4 mod 64 banks)

typedef __attribute__((ext_vector_type(8)))  float  f32x8;
typedef __attribute__((ext_vector_type(16))) __bf16 bf16x16;
typedef __attribute__((ext_vector_type(8)))  __bf16 bf16x8;

// ---------------------------------------------------------------------------
// Prep kernel 1: zero the column-sum workspace (harness poisons ws with 0xAA)
// ---------------------------------------------------------------------------
__global__ void fio_zero_sums(float* __restrict__ sums) {
    sums[blockIdx.x * 256 + threadIdx.x] = 0.0f;   // grid 8 x 256 = 2048 = B*D
}

// ---------------------------------------------------------------------------
// Prep kernel 2: column sums of fields over n (for the lag means)
// ---------------------------------------------------------------------------
__global__ void fio_colsum(const float* __restrict__ fields, float* __restrict__ sums) {
    const int d = blockIdx.x * 256 + threadIdx.x;
    const int b = blockIdx.z;
    const int n0 = blockIdx.y * (NN / 16);
    const float* p = fields + ((size_t)b * NN + n0) * DD + d;
    float s = 0.0f;
    #pragma unroll 4
    for (int n = 0; n < NN / 16; ++n) s += p[(size_t)n * DD];
    atomicAdd(&sums[b * DD + d], s);
}

// ---------------------------------------------------------------------------
// Prep kernel 3: gsel[b][j][d] = (1/3) * sum_{lag=1..j} K[h][lag-1][c] *
//   mean_shift[b][lag][d]; zero for c=d%64>=3.  Row n uses j=min(3,N-1-n).
// ---------------------------------------------------------------------------
__global__ void fio_gsel(const float* __restrict__ fields,
                         const float* __restrict__ ck,      // corr_kernels [8][3][3]
                         const float* __restrict__ sums,
                         float* __restrict__ gsel) {
    const int idx = blockIdx.x * 256 + threadIdx.x;  // 0..2047
    const int b = idx >> 9;
    const int d = idx & (DD - 1);
    const int h = d >> 6;
    const int c = d & 63;
    const float S  = sums[idx];
    const float f0 = fields[((size_t)b * NN + 0) * DD + d];
    const float f1 = fields[((size_t)b * NN + 1) * DD + d];
    const float pre[3] = {0.0f, f0, f0 + f1};
    float acc = 0.0f;
    gsel[((size_t)b * 4 + 0) * DD + d] = 0.0f;
    #pragma unroll
    for (int lag = 1; lag <= 3; ++lag) {
        const float mean = (S - pre[lag - 1]) / (float)(NN - lag);
        const float kw = (c < 3) ? ck[h * 9 + (lag - 1) * 3 + c] : 0.0f;
        acc += kw * mean * (1.0f / 3.0f);
        gsel[((size_t)b * 4 + lag) * DD + d] = acc;
    }
}

// ---------------------------------------------------------------------------
// Prep kernel 4: convert + PACK the four 512x512 f32 weights into the exact
// B-fragment order fragBpk reads (contiguous k, as fragA16x32):  packed[(ctile*16+ks)*512 + lane*16 + e]
// where col = ctile*16 + (lane&15), k = ks*32 + (lane>>4)*16 + e.
// Steady-state B loads become 32 lanes x 32 B fully contiguous (1 KB/instr).
// ---------------------------------------------------------------------------
__global__ void fio_pack_weights(const float* __restrict__ w1, const float* __restrict__ w2,
                                 const float* __restrict__ w3, const float* __restrict__ ow,
                                 __bf16* __restrict__ out) {
    const size_t i = (size_t)blockIdx.x * 256 + threadIdx.x;   // 0..4*262144-1
    const int m = (int)(i >> 18);
    const int r = (int)(i & 262143);
    const int e     = r & 15;
    const int lane  = (r >> 4) & 31;
    const int ks    = (r >> 9) & 15;
    const int ctile = r >> 13;
    const int col = ctile * 16 + (lane & 15);
    const int k   = ks * 32 + (lane >> 4) * 16 + e;
    const size_t off = (size_t)col * DD + k;
    float v;
    if      (m == 0) v = w1[off];
    else if (m == 1) v = w2[off];
    else if (m == 2) v = w3[off];
    else             v = ow[off];
    out[i] = (__bf16)v;
}

// ---------------------------------------------------------------------------
// Fragment loaders (wave32, contiguous k per lane half, same order for A and B)
// A 16x32 bf16 from padded LDS: lane l -> row m=l&15, K={hi*16..hi*16+15}
// B 32x16 bf16 from packed global: lane's 32 B stored contiguously.
// ---------------------------------------------------------------------------
__device__ inline bf16x16 fragA16x32(const __bf16* __restrict__ buf,
                                     int r0, int kk, int lane) {
    const int r = lane & 15, hi = lane >> 4;
    const __bf16* p = buf + (size_t)(r0 + r) * SP + kk + hi * 16;
    bf16x8 lo = *(const bf16x8*)p;
    bf16x8 hv = *(const bf16x8*)(p + 8);
    return __builtin_shufflevector(lo, hv, 0,1,2,3,4,5,6,7,8,9,10,11,12,13,14,15);
}

__device__ inline bf16x16 fragBpk(const __bf16* __restrict__ pkd,
                                  int ctile, int ks, int lane) {
    const __bf16* p = pkd + (((size_t)(ctile * 16 + ks)) << 9) + (lane << 4);
    bf16x8 lo = *(const bf16x8*)p;
    bf16x8 hv = *(const bf16x8*)(p + 8);
    return __builtin_shufflevector(lo, hv, 0,1,2,3,4,5,6,7,8,9,10,11,12,13,14,15);
}

// One GEMM stripe per wave: 32 rows x 64 cols, full K=512 reduction.
// 2 row-blocks x 4 col-tiles = 8 WMMAs per K-step with 8 global + 4 ds b128s.
__device__ inline void fio_gemm(const __bf16* __restrict__ buf,
                                const __bf16* __restrict__ pkdW,
                                int ct0, int lane, f32x8 acc[2][4]) {
    #pragma unroll 2
    for (int ks = 0; ks < 16; ++ks) {
        const bf16x16 a0 = fragA16x32(buf,  0, ks * 32, lane);
        const bf16x16 a1 = fragA16x32(buf, 16, ks * 32, lane);
        #pragma unroll
        for (int t = 0; t < 4; ++t) {
            const bf16x16 bfr = fragBpk(pkdW, ct0 + t, ks, lane);
            acc[0][t] = __builtin_amdgcn_wmma_f32_16x16x32_bf16(
                false, a0, false, bfr, (short)0, acc[0][t], false, false);
            acc[1][t] = __builtin_amdgcn_wmma_f32_16x16x32_bf16(
                false, a1, false, bfr, (short)0, acc[1][t], false, false);
        }
    }
}

// ---------------------------------------------------------------------------
// Fused kernel: 4 chained GEMMs per 32-row tile, activations ping-pong in LDS
// (bf16, padded stride), weights streamed pre-packed from L2, f32 accumulate.
//   stage1: h1 = relu(x  @ W1^T + b1)
//   stage2: h2 = relu(h1 @ W2^T + b2)
//   stage3: nl = h2 @ W3^T + b3 ; e = f*(1 + iw0*gsel) + iw1*nl
//   stage4: out = e @ OW^T + out_b   (f32 to HBM)
// ---------------------------------------------------------------------------
__global__ __launch_bounds__(256)
void fio_fused(const float* __restrict__ fields,
               const float* __restrict__ b1, const float* __restrict__ b2,
               const float* __restrict__ b3, const float* __restrict__ outb,
               const float* __restrict__ gsel,
               const __bf16* __restrict__ w1p, const __bf16* __restrict__ w2p,
               const __bf16* __restrict__ w3p, const __bf16* __restrict__ owp,
               const float* __restrict__ interw,
               float* __restrict__ out) {
    __shared__ __align__(16) __bf16 bufA[MT * SP];   // ~32.5 KB
    __shared__ __align__(16) __bf16 bufB[MT * SP];   // ~32.5 KB

    const int tid  = threadIdx.x;
    const int wave = tid >> 5;
    const int lane = tid & 31;
    const int c0  = wave * 64;             // 64-col stripe per wave
    const int ct0 = wave * 4;              // first 16-col tile index

    const int tile = blockIdx.x;           // 1024 tiles
    const int b  = tile >> 8;              // 256 tiles per batch
    const int n0 = (tile & 255) * MT;
    const size_t rowBase = ((size_t)b * NN + n0) * DD;

    // softmax of the 2-element inter_w (scalar, every thread)
    const float e0 = expf(interw[0]);
    const float e1 = expf(interw[1]);
    const float iw0 = e0 / (e0 + e1);
    const float iw1 = e1 / (e0 + e1);

    // ---- stage 0: load 32x512 f32 tile, convert to bf16 in padded LDS ----
    {
        const float4* src = (const float4*)(fields + rowBase);
        for (int i = tid; i < MT * (DD / 4); i += 256) {
            const float4 v = src[i];
            const int row = i >> 7;
            const int o = row * SP + (i & 127) * 4;
            bufA[o + 0] = (__bf16)v.x;
            bufA[o + 1] = (__bf16)v.y;
            bufA[o + 2] = (__bf16)v.z;
            bufA[o + 3] = (__bf16)v.w;
        }
    }
    __syncthreads();

    const int hi = lane >> 4, ln = lane & 15;

    // ---- stage 1: h1 = relu(x @ W1^T + b1) -> bufB ----
    {
        f32x8 acc[2][4] = {};
        fio_gemm(bufA, w1p, ct0, lane, acc);
        #pragma unroll
        for (int rb = 0; rb < 2; ++rb) {
            #pragma unroll
            for (int t = 0; t < 4; ++t) {
                const int col = c0 + t * 16 + ln;
                const float bv = b1[col];
                #pragma unroll
                for (int j = 0; j < 8; ++j) {
                    const int m = rb * 16 + (hi << 3) + j;
                    float v = acc[rb][t][j] + bv;
                    v = v > 0.0f ? v : 0.0f;
                    bufB[m * SP + col] = (__bf16)v;
                }
            }
        }
    }
    __syncthreads();

    // ---- stage 2: h2 = relu(h1 @ W2^T + b2) -> bufA ----
    {
        f32x8 acc[2][4] = {};
        fio_gemm(bufB, w2p, ct0, lane, acc);
        #pragma unroll
        for (int rb = 0; rb < 2; ++rb) {
            #pragma unroll
            for (int t = 0; t < 4; ++t) {
                const int col = c0 + t * 16 + ln;
                const float bv = b2[col];
                #pragma unroll
                for (int j = 0; j < 8; ++j) {
                    const int m = rb * 16 + (hi << 3) + j;
                    float v = acc[rb][t][j] + bv;
                    v = v > 0.0f ? v : 0.0f;
                    bufA[m * SP + col] = (__bf16)v;
                }
            }
        }
    }
    __syncthreads();

    // ---- stage 3: nl = h2 @ W3^T + b3; enhanced -> bufB ----
    {
        f32x8 acc[2][4] = {};
        fio_gemm(bufA, w3p, ct0, lane, acc);
        #pragma unroll
        for (int rb = 0; rb < 2; ++rb) {
            #pragma unroll
            for (int t = 0; t < 4; ++t) {
                const int col = c0 + t * 16 + ln;
                const float bv = b3[col];
                const float gs3 = gsel[((size_t)b * 4 + 3) * DD + col];
                #pragma unroll
                for (int j = 0; j < 8; ++j) {
                    const int m = rb * 16 + (hi << 3) + j;
                    const int n = n0 + m;
                    const float nl = acc[rb][t][j] + bv;
                    const float fv = fields[rowBase + (size_t)m * DD + col];
                    int jn = NN - 1 - n;  jn = jn > 3 ? 3 : jn;   // edge rows
                    const float gs = (jn == 3) ? gs3
                                   : gsel[((size_t)b * 4 + jn) * DD + col];
                    const float e = fv * (1.0f + iw0 * gs) + iw1 * nl;
                    bufB[m * SP + col] = (__bf16)e;
                }
            }
        }
    }
    __syncthreads();

    // ---- stage 4: out = enhanced @ OW^T + out_b (f32 to HBM) ----
    {
        f32x8 acc[2][4] = {};
        fio_gemm(bufB, owp, ct0, lane, acc);
        #pragma unroll
        for (int rb = 0; rb < 2; ++rb) {
            #pragma unroll
            for (int t = 0; t < 4; ++t) {
                const int col = c0 + t * 16 + ln;
                const float bv = outb[col];
                #pragma unroll
                for (int j = 0; j < 8; ++j) {
                    const int m = rb * 16 + (hi << 3) + j;
                    out[rowBase + (size_t)m * DD + col] = acc[rb][t][j] + bv;
                }
            }
        }
    }
}

// ---------------------------------------------------------------------------
// Launch
// ---------------------------------------------------------------------------
extern "C" void kernel_launch(void* const* d_in, const int* in_sizes, int n_in,
                              void* d_out, int out_size, void* d_ws, size_t ws_size,
                              hipStream_t stream) {
    (void)in_sizes; (void)n_in; (void)out_size; (void)ws_size;

    const float* fields = (const float*)d_in[0];
    const float* ck     = (const float*)d_in[1];
    const float* w1     = (const float*)d_in[2];
    const float* b1     = (const float*)d_in[3];
    const float* w2     = (const float*)d_in[4];
    const float* b2     = (const float*)d_in[5];
    const float* w3     = (const float*)d_in[6];
    const float* b3     = (const float*)d_in[7];
    const float* ow     = (const float*)d_in[8];
    const float* ob     = (const float*)d_in[9];
    const float* iw     = (const float*)d_in[10];
    float* out          = (float*)d_out;

    // workspace layout: sums (8KB) | gsel (32KB) | packed bf16 weights (2MB)
    char* ws = (char*)d_ws;
    float*  sums = (float*)ws;                          // B*D floats
    float*  gsel = (float*)(ws + 8192);                 // B*4*D floats
    __bf16* wpk  = (__bf16*)(ws + 8192 + 32768);        // 4 * 512*512 bf16
    __bf16* w1p = wpk;
    __bf16* w2p = wpk + 262144;
    __bf16* w3p = wpk + 2 * 262144;
    __bf16* owp = wpk + 3 * 262144;

    fio_zero_sums<<<8, 256, 0, stream>>>(sums);
    fio_colsum<<<dim3(DD / 256, 16, BB), 256, 0, stream>>>(fields, sums);
    fio_gsel<<<8, 256, 0, stream>>>(fields, ck, sums, gsel);
    fio_pack_weights<<<4096, 256, 0, stream>>>(w1, w2, w3, ow, wpk);
    fio_fused<<<(BB * NN) / MT, 256, 0, stream>>>(fields, b1, b2, b3, ob, gsel,
                                                  w1p, w2p, w3p, owp, iw, out);
}
